// TensorEmbedding_33457795236216
// MI455X (gfx1250) — matrix operand order repaired and checked by reading, not compiled, not run on hardware
//
#include <hip/hip_runtime.h>
#include <cstddef>

typedef __attribute__((ext_vector_type(16))) _Float16     v16h;
typedef __attribute__((ext_vector_type(8)))  float        v8f;
typedef __attribute__((ext_vector_type(4)))  unsigned int v4u;

#define HC 64
#define RCUT 4.5f
#define PI_F 3.14159265358979323846f

__device__ __forceinline__ v8f wmma16(v16h a, v16h b, v8f c) {
    return __builtin_amdgcn_wmma_f32_16x16x32_f16(false, a, false, b, (short)0, c, false, false);
}

__device__ __forceinline__ unsigned int pack2h(float a, float b) {
    union { _Float16 h[2]; unsigned int u; } t;
    t.h[0] = (_Float16)a; t.h[1] = (_Float16)b;
    return t.u;
}

__device__ __forceinline__ void atomAddF(float* p, float v) {
    __hip_atomic_fetch_add(p, v, __ATOMIC_RELAXED, __HIP_MEMORY_SCOPE_AGENT);
}

// Read a pre-swizzled B-fragment: 8 contiguous dwords per lane -> 2x ds_load_b128.
__device__ __forceinline__ v16h loadBfrag(const unsigned int* basep, int idx) {
    union { v4u u2[2]; v16h h; } t;
    const v4u* p = (const v4u*)(basep + idx * 8);
    t.u2[0] = p[0];
    t.u2[1] = p[1];
    return t.h;
}

// ---------------------------------------------------------------------------
// Edge kernel: per 16-edge tile, WMMA GEMMs for Zij (K=128) and g1..g3 (K=64),
// then 10-component atomic scatter into acc[node][h][10].
// Weight B-matrices live in LDS pre-swizzled into WMMA fragment order:
//   index = ((ks*2 + half)*64 + h)*8 + v ,  dword = pack(B[k][h], B[k+1][h]),
//   k = ks*32 + half*16 + 2v.
// ---------------------------------------------------------------------------
__global__ __launch_bounds__(256) void tensoremb_edge_kernel(
    const int* __restrict__ z, const int* __restrict__ ei,
    const float* __restrict__ ew, const float* __restrict__ evn,
    const float* __restrict__ eattr, const float* __restrict__ emb,
    const float* __restrict__ emb2_w, const float* __restrict__ emb2_b,
    const float* __restrict__ dp1_w, const float* __restrict__ dp1_b,
    const float* __restrict__ dp2_w, const float* __restrict__ dp2_b,
    const float* __restrict__ dp3_w, const float* __restrict__ dp3_b,
    float* __restrict__ acc, int E_)
{
    __shared__ unsigned int sBfE[4 * 2 * 64 * 8];      // emb2: ks(4) x half(2) x h(64) x v(8)
    __shared__ unsigned int sBfD[3][2 * 2 * 64 * 8];   // dpX : ks(2) x half(2) x h(64) x v(8)

    const int tid = threadIdx.x;
    for (int i = tid; i < 4096; i += 256) {
        int v = i & 7, h = (i >> 3) & 63, hf = (i >> 9) & 1, ks = i >> 10;
        int k = ks * 32 + hf * 16 + 2 * v;
        sBfE[i] = pack2h(emb2_w[h * 128 + k], emb2_w[h * 128 + k + 1]);
    }
    for (int i = tid; i < 3 * 2048; i += 256) {
        int wg = i >> 11;
        int r = i & 2047;
        int v = r & 7, h = (r >> 3) & 63, hf = (r >> 9) & 1, ks = r >> 10;
        int k = ks * 32 + hf * 16 + 2 * v;
        const float* W = (wg == 0) ? dp1_w : ((wg == 1) ? dp2_w : dp3_w);
        sBfD[wg][r] = pack2h(W[h * 64 + k], W[h * 64 + k + 1]);
    }
    __syncthreads();

    const int wave = tid >> 5;
    const int lane = tid & 31;
    const int half = lane >> 4;
    const int mrow = lane & 15;
    const int etiles = (E_ + 15) >> 4;
    const int tile = blockIdx.x * 8 + wave;
    if (tile >= etiles) return;
    const int ebase = tile * 16;

    // Per-lane edge metadata for the 8 C-rows this lane owns (m = r + 8*half).
    // Padded edges get Cc = 0 -> all scattered contributions are exactly 0.0f,
    // so the atomics below can run unconditionally (no exec-mask churn).
    int   srcIdx[8];
    float Cc[8], vx[8], vy[8], vz[8];
#pragma unroll
    for (int r = 0; r < 8; ++r) {
        int m = r + 8 * half;
        int e = ebase + m;
        bool ok = (e < E_);
        int ec = ok ? e : 0;
        srcIdx[r] = ei[ec];
        float w = ew[ec];
        Cc[r] = ok ? 0.5f * (__cosf(w * (PI_F / RCUT)) + 1.0f) * (w < RCUT ? 1.0f : 0.0f) : 0.0f;
        vx[r] = evn[ec * 3 + 0];
        vy[r] = evn[ec * 3 + 1];
        vz[r] = evn[ec * 3 + 2];
    }

    // A-fragments for this lane's A-row (m = mrow).
    int eA = ebase + mrow; if (eA >= E_) eA = E_ - 1;
    const float* rowS = emb + (size_t)z[ei[eA]] * HC;
    const float* rowD = emb + (size_t)z[ei[E_ + eA]] * HC;
    const float* rowAttr = eattr + (size_t)eA * HC;

    v16h az[4];   // Zcat 16x128
#pragma unroll
    for (int ks = 0; ks < 4; ++ks)
#pragma unroll
        for (int v = 0; v < 8; ++v) {
            int kl = (half << 4) + (v << 1);
            int k = ks * 32 + kl;
            float f0 = (k < 64) ? rowS[k] : rowD[k - 64];
            float f1 = (k < 64) ? rowS[k + 1] : rowD[k + 1 - 64];
            az[ks][2 * v] = (_Float16)f0; az[ks][2 * v + 1] = (_Float16)f1;
        }
    v16h ae[2];   // edge_attr 16x64
#pragma unroll
    for (int ks = 0; ks < 2; ++ks)
#pragma unroll
        for (int v = 0; v < 8; ++v) {
            int kl = (half << 4) + (v << 1);
            int k = ks * 32 + kl;
            ae[ks][2 * v] = (_Float16)rowAttr[k];
            ae[ks][2 * v + 1] = (_Float16)rowAttr[k + 1];
        }

    for (int ntile = 0; ntile < 4; ++ntile) {
        const int h = ntile * 16 + mrow;
        v8f cz = {};
#pragma unroll
        for (int ks = 0; ks < 4; ++ks)
            cz = wmma16(az[ks], loadBfrag(sBfE, (ks * 2 + half) * 64 + h), cz);
        v8f c1 = {}, c2 = {}, c3 = {};
#pragma unroll
        for (int ks = 0; ks < 2; ++ks) {
            int fidx = (ks * 2 + half) * 64 + h;
            c1 = wmma16(ae[ks], loadBfrag(sBfD[0], fidx), c1);
            c2 = wmma16(ae[ks], loadBfrag(sBfD[1], fidx), c2);
            c3 = wmma16(ae[ks], loadBfrag(sBfD[2], fidx), c3);
        }
        const float bz = emb2_b[h], b1v = dp1_b[h], b2v = dp2_b[h], b3v = dp3_b[h];
#pragma unroll
        for (int r = 0; r < 8; ++r) {
            float cw = Cc[r];
            float Z  = cz[r] + bz;
            float g1 = (c1[r] + b1v) * cw * Z;
            float g2 = (c2[r] + b2v) * cw * Z;
            float g3 = (c3[r] + b3v) * cw * Z;
            float x = vx[r], y = vy[r], zc = vz[r];
            float* p = acc + ((size_t)srcIdx[r] * HC + h) * 10;
            atomAddF(p + 0, g1);
            atomAddF(p + 1, g2 * x);
            atomAddF(p + 2, g2 * y);
            atomAddF(p + 3, g2 * zc);
            atomAddF(p + 4, g3 * (x * x - (1.0f / 3.0f)));
            atomAddF(p + 5, g3 * (x * y));
            atomAddF(p + 6, g3 * (x * zc));
            atomAddF(p + 7, g3 * (y * y - (1.0f / 3.0f)));
            atomAddF(p + 8, g3 * (y * zc));
            atomAddF(p + 9, g3 * (zc * zc - (1.0f / 3.0f)));
        }
    }
}

// ---------------------------------------------------------------------------
// Node kernel: tn -> LayerNorm -> MLP1 -> MLP2(rows 3k) -> u/d channel mixes,
// all via WMMA on 16-node tiles; output is diagonal 3x3 per (node, k).
// ---------------------------------------------------------------------------
__global__ __launch_bounds__(128) void tensoremb_node_kernel(
    const float* __restrict__ acc,
    const float* __restrict__ ln_g, const float* __restrict__ ln_b,
    const float* __restrict__ w1, const float* __restrict__ b1,
    const float* __restrict__ w2, const float* __restrict__ b2,
    const float* __restrict__ m0, const float* __restrict__ m1,
    const float* __restrict__ m2,
    float* __restrict__ out, int N_)
{
    __shared__ float    sT[4][16][64];     // tn, later reused for u
    __shared__ _Float16 sH1[4][16][128];   // MLP1 activations
    __shared__ float    sN0[4][16][64];    // n0
    __shared__ float    sMu[4][16], sRs[4][16];

    const int tid = threadIdx.x;
    const int w = tid >> 5, lane = tid & 31;
    const int half = lane >> 4, mrow = lane & 15;
    const int tile = blockIdx.x * 4 + w;
    const int base = tile * 16;
    const bool act = (base < N_);

    // 1) Frobenius norm^2 of T = I*eye + skew(a) + sym(s), per (node, h)
    for (int i = lane; i < 16 * 64; i += 32) {
        int m = i >> 6, h = i & 63;
        int n = base + m;
        float tn = 0.0f;
        if (act && n < N_) {
            const float* p = acc + ((size_t)n * HC + h) * 10;
            float I = p[0], ax = p[1], ay = p[2], azv = p[3];
            float sxx = p[4], sxy = p[5], sxz = p[6], syy = p[7], syz = p[8], szz = p[9];
            float t00 = I + sxx, t11 = I + syy, t22 = I + szz;
            float t01 = sxy - azv, t10 = sxy + azv;
            float t02 = sxz + ay,  t20 = sxz - ay;
            float t12 = syz - ax,  t21 = syz + ax;
            tn = t00 * t00 + t11 * t11 + t22 * t22 + t01 * t01 + t10 * t10 +
                 t02 * t02 + t20 * t20 + t12 * t12 + t21 * t21;
        }
        sT[w][m][h] = tn;
    }
    __syncthreads();

    // 2) LayerNorm stats per node
    if (lane < 16) {
        float s = 0.0f, s2 = 0.0f;
        for (int h = 0; h < 64; ++h) { float v = sT[w][lane][h]; s += v; s2 += v * v; }
        float mu = s * (1.0f / 64.0f);
        float var = s2 * (1.0f / 64.0f) - mu * mu;
        sMu[w][lane] = mu;
        sRs[w][lane] = rsqrtf(var + 1e-5f);
    }
    __syncthreads();

    // 3) MLP1: [16,64] x w1^T -> [16,128], silu
    v16h aL[2];
    {
        float mu = sMu[w][mrow], rs = sRs[w][mrow];
#pragma unroll
        for (int ks = 0; ks < 2; ++ks)
#pragma unroll
            for (int v = 0; v < 8; ++v) {
                int kl = (half << 4) + (v << 1);
                int k = ks * 32 + kl;
                float x0 = (sT[w][mrow][k] - mu) * rs * ln_g[k] + ln_b[k];
                float x1 = (sT[w][mrow][k + 1] - mu) * rs * ln_g[k + 1] + ln_b[k + 1];
                aL[ks][2 * v] = (_Float16)x0; aL[ks][2 * v + 1] = (_Float16)x1;
            }
    }
    for (int nt = 0; nt < 8; ++nt) {
        int n = nt * 16 + mrow;
        v8f c = {};
#pragma unroll
        for (int ks = 0; ks < 2; ++ks) {
            v16h bfr;
#pragma unroll
            for (int v = 0; v < 8; ++v) {
                int k = ks * 32 + half * 16 + 2 * v;
                bfr[2 * v]     = (_Float16)w1[n * 64 + k];
                bfr[2 * v + 1] = (_Float16)w1[n * 64 + k + 1];
            }
            c = wmma16(aL[ks], bfr, c);
        }
        float bb = b1[n];
#pragma unroll
        for (int r = 0; r < 8; ++r) {
            int m = r + 8 * half;
            float x = c[r] + bb;
            sH1[w][m][n] = (_Float16)(x * (1.0f / (1.0f + __expf(-x))));
        }
    }
    __syncthreads();

    // 4) MLP2 (only rows 3k of w2 matter): [16,128] x w2sel^T -> n0[16,64], silu
    v16h a2[4];
#pragma unroll
    for (int ks = 0; ks < 4; ++ks)
#pragma unroll
        for (int v = 0; v < 8; ++v) {
            int kl = (half << 4) + (v << 1);
            int k = ks * 32 + kl;
            a2[ks][2 * v] = sH1[w][mrow][k];
            a2[ks][2 * v + 1] = sH1[w][mrow][k + 1];
        }
    for (int nt = 0; nt < 4; ++nt) {
        int n = nt * 16 + mrow;
        v8f c = {};
#pragma unroll
        for (int ks = 0; ks < 4; ++ks) {
            v16h bfr;
#pragma unroll
            for (int v = 0; v < 8; ++v) {
                int k = ks * 32 + half * 16 + 2 * v;
                bfr[2 * v]     = (_Float16)w2[(size_t)(3 * n) * 128 + k];
                bfr[2 * v + 1] = (_Float16)w2[(size_t)(3 * n) * 128 + k + 1];
            }
            c = wmma16(a2[ks], bfr, c);
        }
        float bb = b2[3 * n];
#pragma unroll
        for (int r = 0; r < 8; ++r) {
            int m = r + 8 * half;
            float x = c[r] + bb;
            sN0[w][m][n] = x * (1.0f / (1.0f + __expf(-x)));
        }
    }
    __syncthreads();

    // 5) u = (Isum @ m0^T) * n0   (Isum = acc component 0)
    v16h aI[2];
#pragma unroll
    for (int ks = 0; ks < 2; ++ks)
#pragma unroll
        for (int v = 0; v < 8; ++v) {
            int kl = (half << 4) + (v << 1);
            int k = ks * 32 + kl;
            int n = base + mrow;
            float f0 = 0.0f, f1 = 0.0f;
            if (act && n < N_) {
                f0 = acc[((size_t)n * HC + k) * 10];
                f1 = acc[((size_t)n * HC + k + 1) * 10];
            }
            aI[ks][2 * v] = (_Float16)f0; aI[ks][2 * v + 1] = (_Float16)f1;
        }
    for (int nt = 0; nt < 4; ++nt) {
        int n = nt * 16 + mrow;
        v8f c = {};
#pragma unroll
        for (int ks = 0; ks < 2; ++ks) {
            v16h bfr;
#pragma unroll
            for (int v = 0; v < 8; ++v) {
                int k = ks * 32 + half * 16 + 2 * v;
                bfr[2 * v]     = (_Float16)m0[n * 64 + k];
                bfr[2 * v + 1] = (_Float16)m0[n * 64 + k + 1];
            }
            c = wmma16(aI[ks], bfr, c);
        }
#pragma unroll
        for (int r = 0; r < 8; ++r) {
            int m = r + 8 * half;
            sT[w][m][n] = c[r] * sN0[w][m][n];   // reuse sT for u
        }
    }
    __syncthreads();

    // 6) d = u + (u @ (m1+m2)^T) * n0 ; write diagonal 3x3 output
    v16h aU[2];
#pragma unroll
    for (int ks = 0; ks < 2; ++ks)
#pragma unroll
        for (int v = 0; v < 8; ++v) {
            int kl = (half << 4) + (v << 1);
            int k = ks * 32 + kl;
            aU[ks][2 * v] = (_Float16)sT[w][mrow][k];
            aU[ks][2 * v + 1] = (_Float16)sT[w][mrow][k + 1];
        }
    for (int nt = 0; nt < 4; ++nt) {
        int n = nt * 16 + mrow;
        v8f c = {};
#pragma unroll
        for (int ks = 0; ks < 2; ++ks) {
            v16h bfr;
#pragma unroll
            for (int v = 0; v < 8; ++v) {
                int k = ks * 32 + half * 16 + 2 * v;
                bfr[2 * v]     = (_Float16)(m1[n * 64 + k] + m2[n * 64 + k]);
                bfr[2 * v + 1] = (_Float16)(m1[n * 64 + k + 1] + m2[n * 64 + k + 1]);
            }
            c = wmma16(aU[ks], bfr, c);
        }
#pragma unroll
        for (int r = 0; r < 8; ++r) {
            int m = r + 8 * half;
            int node = base + m;
            if (act && node < N_) {
                float d = sT[w][m][n] + c[r] * sN0[w][m][n];
                float* o = out + ((size_t)node * HC + n) * 9;
                o[0] = d;    o[1] = 0.0f; o[2] = 0.0f;
                o[3] = 0.0f; o[4] = d;    o[5] = 0.0f;
                o[6] = 0.0f; o[7] = 0.0f; o[8] = d;
            }
        }
    }
}

extern "C" void kernel_launch(void* const* d_in, const int* in_sizes, int n_in,
                              void* d_out, int out_size, void* d_ws, size_t ws_size,
                              hipStream_t stream) {
    const int*   z      = (const int*)d_in[0];
    const int*   ei     = (const int*)d_in[1];
    const float* ew     = (const float*)d_in[2];
    const float* evn    = (const float*)d_in[3];
    const float* eattr  = (const float*)d_in[4];
    const float* emb    = (const float*)d_in[5];
    const float* emb2_w = (const float*)d_in[6];
    const float* emb2_b = (const float*)d_in[7];
    const float* dp1_w  = (const float*)d_in[8];
    const float* dp1_b  = (const float*)d_in[9];
    const float* dp2_w  = (const float*)d_in[10];
    const float* dp2_b  = (const float*)d_in[11];
    const float* dp3_w  = (const float*)d_in[12];
    const float* dp3_b  = (const float*)d_in[13];
    const float* ln_g   = (const float*)d_in[14];
    const float* ln_b   = (const float*)d_in[15];
    const float* w1     = (const float*)d_in[16];
    const float* b1     = (const float*)d_in[17];
    const float* w2     = (const float*)d_in[18];
    const float* b2     = (const float*)d_in[19];
    const float* m0     = (const float*)d_in[20];
    const float* m1     = (const float*)d_in[21];
    const float* m2     = (const float*)d_in[22];

    const int N_ = in_sizes[0];
    const int E_ = in_sizes[2];

    float* acc = (float*)d_ws;
    size_t accBytes = (size_t)N_ * HC * 10 * sizeof(float);
    hipMemsetAsync(acc, 0, accBytes, stream);

    int etiles = (E_ + 15) / 16;
    int eblocks = (etiles + 7) / 8;
    tensoremb_edge_kernel<<<eblocks, 256, 0, stream>>>(
        z, ei, ew, evn, eattr, emb, emb2_w, emb2_b,
        dp1_w, dp1_b, dp2_w, dp2_b, dp3_w, dp3_b, acc, E_);

    int ntiles = (N_ + 15) / 16;
    int nblocks = (ntiles + 3) / 4;
    tensoremb_node_kernel<<<nblocks, 128, 0, stream>>>(
        acc, ln_g, ln_b, w1, b1, w2, b2, m0, m1, m2, (float*)d_out, N_);
}
